// GraphOrder_84791244357740
// MI455X (gfx1250) — matrix operand order repaired and checked by reading, not compiled, not run on hardware
//
#include <hip/hip_runtime.h>
#include <hip/hip_bf16.h>
#include <math.h>

typedef _Float16 half_t;
typedef __attribute__((ext_vector_type(16))) _Float16 v16h;
typedef __attribute__((ext_vector_type(8)))  float    v8f;

#define N_NODES 131072
#define KCH 2
#define DDIM 256
#define MT 16          // WMMA M tile (nodes per workgroup)
#define KKCH 8         // DDIM / 32 K-chunks per WMMA K-loop
#define NWTILES 64     // 48 W_iou col-tiles + 16 W_f col-tiles
#define NTILES 80      // 48 iou + 16 f0 + 16 f1 output tiles
#define WAVES 8
#define LN_EPS 1e-5f

// --------------------------------------------------------------------------
// Repack fp32 weights into f16 WMMA B-matrix fragments (32x16 per K-chunk).
// B layout (V_WMMA_F32_16X16X32_F16): lanes 0-15 hold K={0..7,16..23} (col N=lane),
// lanes 16-31 hold K={8..15,24..31} (col N=lane-16); 2 halfs/VGPR along K.
// ws layout: [wtile(64)][kk(8)][lane(32)][j(16)] f16  (j = in-lane K order)
// --------------------------------------------------------------------------
__global__ void prep_weights(const float* __restrict__ W_iou,
                             const float* __restrict__ W_f,
                             half_t* __restrict__ wsW) {
  const int tile = blockIdx.x >> 3;
  const int kk   = blockIdx.x & 7;
  const int lane = threadIdx.x;
  half_t* dst = wsW + (((size_t)tile * KKCH + kk) * 32 + lane) * 16;
  const int khi = (lane >= 16) ? 8 : 0;
  const int n   = lane & 15;
  #pragma unroll
  for (int j = 0; j < 16; ++j) {
    const int k = kk * 32 + khi + (j & 7) + ((j >= 8) ? 16 : 0);  // K index (row of B / row of W)
    float v;
    if (tile < 48) v = W_iou[(size_t)k * 768 + tile * 16 + n];
    else           v = W_f  [(size_t)k * 256 + (tile - 48) * 16 + n];
    dst[j] = (half_t)v;
  }
}

// --------------------------------------------------------------------------
// Fused TreeLSTM node-batch kernel: one 16-node tile per workgroup.
// --------------------------------------------------------------------------
__global__ __launch_bounds__(256)
void treelstm_fused(const float* __restrict__ in,    // (N, 2, 2, 256) f32
                    const half_t* __restrict__ wsW,  // weight fragments
                    const float* __restrict__ b_iou, // (768,)
                    const float* __restrict__ b_f,   // (256,)
                    const float* __restrict__ gamma, // (256,)
                    const float* __restrict__ beta,  // (256,)
                    float* __restrict__ out) {       // (N, 2, 256) f32
  // A fragments for hsum, h0, h1 : [sel][kk][lane] one v16h each
  __shared__ v16h  AhV[3][KKCH][32];                 // 24 KB
  __shared__ float ioub[MT][768];                    // 48 KB
  __shared__ float fbuf[2][MT][256];                 // 32 KB

  const int t    = threadIdx.x;
  const int wave = t >> 5;
  const int lane = t & 31;
  const size_t gbase = (size_t)blockIdx.x * MT;
  half_t* Ahh = (half_t*)AhV;

  // ---- stage child_h into LDS as f16 A fragments -------------------------
  // A layout (16-bit A 16x32): lanes 0-15: M=lane, K={0..7,16..23};
  // lanes 16-31: M=lane-16, K={8..15,24..31}; packed 2 halfs/VGPR along K.
  {
    const int dq = (t & 63) * 4;   // dim base, 4 consecutive dims
    const int mq = t >> 6;         // node sub-offset
    #pragma unroll
    for (int it = 0; it < 4; ++it) {
      const int m = it * 4 + mq;
      const size_t g = gbase + m;
      const float4 h0 = *(const float4*)(in + ((g * KCH + 0) * 2 + 0) * DDIM + dq);
      const float4 h1 = *(const float4*)(in + ((g * KCH + 1) * 2 + 0) * DDIM + dq);
      const float h0a[4] = {h0.x, h0.y, h0.z, h0.w};
      const float h1a[4] = {h1.x, h1.y, h1.z, h1.w};
      #pragma unroll
      for (int q = 0; q < 4; ++q) {
        const int d  = dq + q;
        const int kk = d >> 5;
        const int kp = d & 31;
        const int ln = m + ((kp >> 3) & 1) * 16;
        const int j  = (kp & 7) + ((kp >= 16) ? 8 : 0);
        const size_t fi = (((size_t)kk) * 32 + ln) * 16 + j;
        Ahh[0 * KKCH * 32 * 16 + fi] = (half_t)(h0a[q] + h1a[q]);
        Ahh[1 * KKCH * 32 * 16 + fi] = (half_t)h0a[q];
        Ahh[2 * KKCH * 32 * 16 + fi] = (half_t)h1a[q];
      }
    }
  }
  __syncthreads();

  // ---- GEMM phase: 80 16-col output tiles across 8 waves -----------------
  const v16h* BW = (const v16h*)wsW;
  for (int tid = wave; tid < NTILES; tid += WAVES) {
    const int asel  = (tid < 48) ? 0 : ((tid < 64) ? 1 : 2);
    const int wtile = (tid < 64) ? tid : (tid - 16);
    const v16h* Bp = BW + (size_t)wtile * KKCH * 32;
    // prefetch the next weight tile this wave will need
    const int ntid = tid + WAVES;
    if (ntid < NTILES) {
      const int nwt = (ntid < 64) ? ntid : (ntid - 16);
      __builtin_prefetch((const void*)(BW + (size_t)nwt * KKCH * 32 + lane), 0, 1);
    }
    v8f acc = {};
    #pragma unroll
    for (int kk = 0; kk < KKCH; ++kk) {
      v16h a = AhV[asel][kk][lane];
      v16h b = Bp[kk * 32 + lane];
      acc = __builtin_amdgcn_wmma_f32_16x16x32_f16(false, a, false, b,
                                                   (short)0, acc, false, false);
    }
    // C/D layout: lane L: VGPR r -> row r + (L>=16 ? 8 : 0), col = L&15
    const int col = lane & 15;
    const int rb  = (lane >= 16) ? 8 : 0;
    if (tid < 48) {
      #pragma unroll
      for (int r = 0; r < 8; ++r) ioub[r + rb][tid * 16 + col] = acc[r];
    } else if (tid < 64) {
      #pragma unroll
      for (int r = 0; r < 8; ++r) fbuf[0][r + rb][(tid - 48) * 16 + col] = acc[r];
    } else {
      #pragma unroll
      for (int r = 0; r < 8; ++r) fbuf[1][r + rb][(tid - 64) * 16 + col] = acc[r];
    }
  }
  __syncthreads();

  // ---- epilogue: gates, cell update, layernorm, outputs ------------------
  // 8 waves x 2 nodes; each lane owns 8 contiguous dims of one node.
  #pragma unroll
  for (int s = 0; s < 2; ++s) {
    const int m = wave * 2 + s;
    const size_t g = gbase + m;
    const int d0 = lane * 8;
    const float4 c0a = *(const float4*)(in + ((g * KCH + 0) * 2 + 1) * DDIM + d0);
    const float4 c0b = *(const float4*)(in + ((g * KCH + 0) * 2 + 1) * DDIM + d0 + 4);
    const float4 c1a = *(const float4*)(in + ((g * KCH + 1) * 2 + 1) * DDIM + d0);
    const float4 c1b = *(const float4*)(in + ((g * KCH + 1) * 2 + 1) * DDIM + d0 + 4);
    const float cc0[8] = {c0a.x, c0a.y, c0a.z, c0a.w, c0b.x, c0b.y, c0b.z, c0b.w};
    const float cc1[8] = {c1a.x, c1a.y, c1a.z, c1a.w, c1b.x, c1b.y, c1b.z, c1b.w};
    float cv[8], ov[8];
    float s1 = 0.f, s2 = 0.f;
    #pragma unroll
    for (int q = 0; q < 8; ++q) {
      const int d = d0 + q;
      float iv = ioub[m][d]        + b_iou[d];
      float o_ = ioub[m][256 + d]  + b_iou[256 + d];
      float uv = ioub[m][512 + d]  + b_iou[512 + d];
      float f0 = fbuf[0][m][d] + b_f[d];
      float f1 = fbuf[1][m][d] + b_f[d];
      iv = 1.f / (1.f + expf(-iv));
      o_ = 1.f / (1.f + expf(-o_));
      uv = tanhf(uv);
      f0 = 1.f / (1.f + expf(-f0));
      f1 = 1.f / (1.f + expf(-f1));
      const float c = iv * uv + f0 * cc0[q] + f1 * cc1[q];
      cv[q] = c;
      ov[q] = o_;
      s1 += c;
      s2 += c * c;
    }
    // wave32 reduction for layernorm stats
    #pragma unroll
    for (int off = 16; off > 0; off >>= 1) {
      s1 += __shfl_xor(s1, off, 32);
      s2 += __shfl_xor(s2, off, 32);
    }
    const float mu  = s1 * (1.f / 256.f);
    const float var = s2 * (1.f / 256.f) - mu * mu;
    const float rs  = rsqrtf(var + LN_EPS);
    float hv[8], cn[8];
    #pragma unroll
    for (int q = 0; q < 8; ++q) {
      const int d = d0 + q;
      const float c = (cv[q] - mu) * rs * gamma[d] + beta[d];
      cn[q] = c;
      hv[q] = ov[q] * tanhf(c);
    }
    float* oh = out + (g * 2 + 0) * DDIM + d0;
    float* oc = out + (g * 2 + 1) * DDIM + d0;
    *(float4*)(oh)     = make_float4(hv[0], hv[1], hv[2], hv[3]);
    *(float4*)(oh + 4) = make_float4(hv[4], hv[5], hv[6], hv[7]);
    *(float4*)(oc)     = make_float4(cn[0], cn[1], cn[2], cn[3]);
    *(float4*)(oc + 4) = make_float4(cn[4], cn[5], cn[6], cn[7]);
  }
}

extern "C" void kernel_launch(void* const* d_in, const int* in_sizes, int n_in,
                              void* d_out, int out_size, void* d_ws, size_t ws_size,
                              hipStream_t stream) {
  const float* inputs = (const float*)d_in[0];
  const float* W_iou  = (const float*)d_in[1];
  const float* b_iou  = (const float*)d_in[2];
  const float* W_f    = (const float*)d_in[3];
  const float* b_f    = (const float*)d_in[4];
  const float* gamma  = (const float*)d_in[5];
  const float* beta   = (const float*)d_in[6];
  half_t* wsW = (half_t*)d_ws;  // 64*8*32*16 halfs = 512 KB

  prep_weights<<<NWTILES * KKCH, 32, 0, stream>>>(W_iou, W_f, wsW);
  treelstm_fused<<<N_NODES / MT, 256, 0, stream>>>(inputs, wsW, b_iou, b_f,
                                                   gamma, beta, (float*)d_out);
}
